// Generator_9981503996603
// MI455X (gfx1250) — matrix operand order repaired and checked by reading, not compiled, not run on hardware
//
#include <hip/hip_runtime.h>
#include <hip/hip_bf16.h>
#include <math.h>

// ============================================================================
// StyleGAN2/DragGAN generator for MI455X (gfx1250, wave32, WMMA).
//
// Modulation folded into activations (B-tile), demodulation into the epilogue
// -> conv weights are batch-shared, every conv is an implicit GEMM
// (M=cout, K=cin*9, N=B*H*W) on v_wmma_f32_16x16x32_f16. Weights are
// pre-packed to f16 [cout, K] per layer so the A-tile stages into LDS via
// GLOBAL_LOAD_ASYNC_TO_LDS_B128 (ASYNCcnt) when the builtin is available.
// Block tile 64x64, 8 waves (4Mx2N), 2 WMMAs per wave per K-step.
// ============================================================================

typedef _Float16 v16h __attribute__((ext_vector_type(16)));
typedef float    v8f  __attribute__((ext_vector_type(8)));
typedef int      v4i  __attribute__((vector_size(16)));   // async-LDS builtin operand

#define SQRT2F 1.41421356237309515f

#if defined(__gfx1250__) && \
    __has_builtin(__builtin_amdgcn_global_load_async_to_lds_b128) && \
    __has_builtin(__builtin_amdgcn_s_wait_asynccnt)
#define USE_ASYNC_LDS 1
#else
#define USE_ASYNC_LDS 0
#endif

__device__ __forceinline__ float lrelu_g(float v) {
    return (v > 0.f ? v : 0.2f * v) * SQRT2F;
}

// ---------------------------------------------------------------------------
// PixelNorm: x = z * rsqrt(mean(z^2) + 1e-8), per row of [8,512]
// ---------------------------------------------------------------------------
__global__ __launch_bounds__(512)
void k_pixelnorm(const float* __restrict__ z, float* __restrict__ out) {
    __shared__ float red[16];
    const int b = blockIdx.x;
    const int t = threadIdx.x;
    float v  = z[b * 512 + t];
    float sq = v * v;
    #pragma unroll
    for (int o = 16; o > 0; o >>= 1) sq += __shfl_down(sq, o, 32);
    if ((t & 31) == 0) red[t >> 5] = sq;
    __syncthreads();
    if (t == 0) {
        float s = 0.f;
        #pragma unroll
        for (int i = 0; i < 16; ++i) s += red[i];
        red[0] = rsqrtf(s * (1.f / 512.f) + 1e-8f);
    }
    __syncthreads();
    out[b * 512 + t] = v * red[0];
}

// ---------------------------------------------------------------------------
// One mapping layer: Y = lrelu( X @ (W^T * 0.01/sqrt(512)) + b*0.01 ) * sqrt2
// X: [8,512] padded to one 16-row WMMA tile; 16 waves x 32 output columns.
// ---------------------------------------------------------------------------
__global__ __launch_bounds__(512)
void k_map_layer(const float* __restrict__ xin, const float* __restrict__ wmat,
                 const float* __restrict__ bias, float* __restrict__ xout) {
    __shared__ _Float16 X[16][520];
    const int tid = threadIdx.x;
    for (int e = tid; e < 16 * 512; e += 512) {
        int r = e >> 9, c = e & 511;
        X[r][c] = (_Float16)(r < 8 ? xin[r * 512 + c] : 0.f);
    }
    __syncthreads();

    const int lane = tid & 31, wave = tid >> 5;
    const int n16 = lane & 15, hh = lane >> 4;
    const int nBase = wave * 32;
    const float wscale = 0.01f * rsqrtf(512.f);   // LR_MUL / sqrt(SD)

    v8f acc0 = {}, acc1 = {};
    for (int kt = 0; kt < 512; kt += 32) {
        v16h af, b0, b1;
        #pragma unroll
        for (int j = 0; j < 8; ++j) {
            int ka = ((j >> 2) << 4) + (hh << 3) + ((j & 3) << 1);
            af[2 * j]     = X[n16][kt + ka];
            af[2 * j + 1] = X[n16][kt + ka + 1];
            int kb = ((j >> 2) << 4) + (hh << 3) + ((j & 3) << 1);
            int k  = kt + kb;
            int c0 = nBase + n16, c1 = c0 + 16;
            b0[2 * j]     = (_Float16)(wmat[c0 * 512 + k]     * wscale);
            b0[2 * j + 1] = (_Float16)(wmat[c0 * 512 + k + 1] * wscale);
            b1[2 * j]     = (_Float16)(wmat[c1 * 512 + k]     * wscale);
            b1[2 * j + 1] = (_Float16)(wmat[c1 * 512 + k + 1] * wscale);
        }
        acc0 = __builtin_amdgcn_wmma_f32_16x16x32_f16(false, af, false, b0,
                                                      (short)0, acc0, false, false);
        acc1 = __builtin_amdgcn_wmma_f32_16x16x32_f16(false, af, false, b1,
                                                      (short)0, acc1, false, false);
    }
    #pragma unroll
    for (int r = 0; r < 8; ++r) {
        int m = hh * 8 + r;
        if (m < 8) {
            int c0 = nBase + n16, c1 = c0 + 16;
            xout[m * 512 + c0] = lrelu_g(acc0[r] + bias[c0] * 0.01f);
            xout[m * 512 + c1] = lrelu_g(acc1[r] + bias[c1] * 0.01f);
        }
    }
}

// ---------------------------------------------------------------------------
// Per-layer style: s[b,ci] = (latent[b] . mw[ci]) / sqrt(512) + mb[ci]
// ---------------------------------------------------------------------------
__global__ void k_style(const float* __restrict__ lat, const float* __restrict__ mw,
                        const float* __restrict__ mb, float* __restrict__ s, int cin) {
    int i = blockIdx.x * blockDim.x + threadIdx.x;
    if (i >= 8 * cin) return;
    int b = i / cin, c = i % cin;
    float acc = 0.f;
    for (int k = 0; k < 512; ++k) acc += lat[b * 512 + k] * mw[c * 512 + k];
    s[i] = acc * rsqrtf(512.f) + mb[c];
}

// ---------------------------------------------------------------------------
// demod[b,co] = rsqrt( (1/(cin*9)) * sum_{ci,k} w^2 s^2 + 1e-8 )
// ---------------------------------------------------------------------------
__global__ void k_demod(const float* __restrict__ w, const float* __restrict__ s,
                        float* __restrict__ demod, int cin, int cout) {
    int i = blockIdx.x * blockDim.x + threadIdx.x;
    if (i >= 8 * cout) return;
    int b = i / cout, co = i % cout;
    float acc = 0.f;
    for (int ci = 0; ci < cin; ++ci) {
        float ss = s[b * cin + ci]; ss *= ss;
        const float* wp = &w[(co * cin + ci) * 9];
        float t = 0.f;
        #pragma unroll
        for (int q = 0; q < 9; ++q) t += wp[q] * wp[q];
        acc += t * ss;
    }
    demod[i] = rsqrtf(acc / (float)(cin * 9) + 1e-8f);
}

// ---------------------------------------------------------------------------
// Pack conv weights to f16 GEMM-A layout [cout, cin*9], scale folded,
// kernel optionally flipped (transpose-conv path). Enables verbatim
// async global->LDS staging of A tiles in the GEMM.
// ---------------------------------------------------------------------------
__global__ void k_pack_w(const float* __restrict__ w, _Float16* __restrict__ pw,
                         int cin, int cout, int flip) {
    const int K = cin * 9;
    int i = blockIdx.x * blockDim.x + threadIdx.x;
    if (i >= cout * K) return;
    int m = i / K, k = i % K;
    int ci = k / 9, r = k % 9;
    int dy = r / 3, dx = r % 3;
    if (flip) { dy = 2 - dy; dx = 2 - dx; }
    pw[i] = (_Float16)(rsqrtf((float)K) * w[((m * cin) + ci) * 9 + dy * 3 + dx]);
}

// ---------------------------------------------------------------------------
// Implicit-GEMM modulated 3x3 conv with WMMA.
// Block tile 64(M) x 64(N); 8 waves as 4(M)x2(N); each wave holds two 16x16
// accumulators -> 2 v_wmma per K-step per wave. A staged async into LDS.
// upsample=1 -> dilate-2/pad-2 transpose conv coords (weights pre-flipped).
// fuseAct=1  -> epilogue + noise + bias + lrelu*sqrt2.
// ---------------------------------------------------------------------------
__global__ __launch_bounds__(256)
void k_conv_wmma(const float* __restrict__ x, const _Float16* __restrict__ pw,
                 const float* __restrict__ s, const float* __restrict__ demod,
                 const float* __restrict__ noise, const float* __restrict__ nw,
                 const float* __restrict__ ab, float* __restrict__ out,
                 int cin, int cout, int H, int W, int OH, int OW,
                 int xBatchStride, int upsample, int fuseAct) {
    const int K    = cin * 9;
    const int Ntot = 8 * OH * OW;
    const int mTile = blockIdx.x * 64;
    const int nTile = blockIdx.y * 64;

    __shared__ _Float16 As[64][40];   // 64 rows x 32 K (row = 64B contiguous)
    __shared__ _Float16 Bs[32][72];   // 32 K x 64 N

    const int tid  = threadIdx.x;
    const int lane = tid & 31;
    const int wave = tid >> 5;
    const int wm = wave & 3;          // M sub-tile (16 rows)
    const int wn = wave >> 2;         // N half (32 cols -> two 16-col tiles)
    const int l16 = lane & 15;
    const int hh  = lane >> 4;

    // async A staging: each thread owns one 16B segment (row = tid/4, seg = tid%4)
    const int arow = tid >> 2;
    const int aseg = tid & 3;

    v8f acc0 = {}, acc1 = {};

    for (int kt = 0; kt < K; kt += 32) {
        // ---- stage A tile: packed f16 weights, verbatim copy ----
#if USE_ASYNC_LDS
        {
            const _Float16* gsrc = &pw[(size_t)(mTile + arow) * K + kt + aseg * 8];
            _Float16* ldst = &As[arow][aseg * 8];
            __builtin_amdgcn_global_load_async_to_lds_b128(
                (v4i*)gsrc, (v4i*)ldst, 0, 0);
        }
#else
        for (int e = tid; e < 64 * 32; e += 256) {
            int mi = e >> 5, ki = e & 31;
            As[mi][ki] = pw[(size_t)(mTile + mi) * K + kt + ki];
        }
#endif
        // ---- stage B tile (im2col of modulated activations, f32->f16) ----
        for (int e = tid; e < 32 * 64; e += 256) {
            int ki = e >> 6, ni = e & 63;
            int k = kt + ki, n = nTile + ni;
            float v = 0.f;
            if (n < Ntot) {
                int ci = k / 9, r = k % 9;
                int dy = r / 3, dx = r % 3;
                int b = n / (OH * OW), rem = n % (OH * OW);
                int oy = rem / OW, ox = rem % OW;
                if (!upsample) {
                    int iy = oy + dy - 1, ix = ox + dx - 1;
                    if (iy >= 0 && iy < H && ix >= 0 && ix < W)
                        v = s[b * cin + ci] *
                            x[b * xBatchStride + (ci * H + iy) * W + ix];
                } else {
                    int ty = oy + dy - 2, tx = ox + dx - 2;   // dilated coords
                    if (ty >= 0 && tx >= 0 && !(ty & 1) && !(tx & 1)) {
                        int iy = ty >> 1, ix = tx >> 1;
                        if (iy < H && ix < W)
                            v = s[b * cin + ci] *
                                x[b * xBatchStride + (ci * H + iy) * W + ix];
                    }
                }
            }
            Bs[ki][ni] = (_Float16)v;
        }
#if USE_ASYNC_LDS
        __builtin_amdgcn_s_wait_asynccnt(0);
#endif
        __syncthreads();

        // ---- fragments per CDNA5 16-bit WMMA VGPR layouts ----
        v16h af, bf0, bf1;
        #pragma unroll
        for (int j = 0; j < 8; ++j) {
            int ka = ((j >> 2) << 4) + (hh << 3) + ((j & 3) << 1);
            af[2 * j]     = As[wm * 16 + l16][ka];
            af[2 * j + 1] = As[wm * 16 + l16][ka + 1];
            int kb = ((j >> 2) << 4) + (hh << 3) + ((j & 3) << 1);
            bf0[2 * j]     = Bs[kb][wn * 32 + l16];
            bf0[2 * j + 1] = Bs[kb + 1][wn * 32 + l16];
            bf1[2 * j]     = Bs[kb][wn * 32 + 16 + l16];
            bf1[2 * j + 1] = Bs[kb + 1][wn * 32 + 16 + l16];
        }
        acc0 = __builtin_amdgcn_wmma_f32_16x16x32_f16(false, af, false, bf0,
                                                      (short)0, acc0, false, false);
        acc1 = __builtin_amdgcn_wmma_f32_16x16x32_f16(false, af, false, bf1,
                                                      (short)0, acc1, false, false);
        __syncthreads();
    }

    // ---- epilogue: demod (+ optional noise/bias/lrelu), two N sub-tiles ----
    #pragma unroll
    for (int half = 0; half < 2; ++half) {
        const int n = nTile + wn * 32 + half * 16 + l16;
        if (n >= Ntot) continue;
        int b = n / (OH * OW), rem = n % (OH * OW);
        int oy = rem / OW, ox = rem % OW;
        float nz = 0.f;
        if (fuseAct) nz = nw[0] * noise[(b * OH + oy) * OW + ox];
        #pragma unroll
        for (int r = 0; r < 8; ++r) {
            int m = mTile + wm * 16 + hh * 8 + r;
            float v = demod[b * cout + m] * (half ? acc1[r] : acc0[r]);
            if (fuseAct) v = lrelu_g(v + nz + ab[m]);
            out[(b * cout + m) * (OH * OW) + rem] = v;
        }
    }
}

// ---------------------------------------------------------------------------
// Post-upsample FIR blur (4x4 separable, x4) + noise + bias + lrelu.
// T: [8,C,OH+1,OW+1] raw transpose-conv output -> out: [8,C,OH,OW]
// ---------------------------------------------------------------------------
__global__ void k_blur_act(const float* __restrict__ T, const float* __restrict__ noise,
                           const float* __restrict__ nw, const float* __restrict__ ab,
                           float* __restrict__ out, int C, int OH, int OW) {
    int i = blockIdx.x * blockDim.x + threadIdx.x;
    int tot = 8 * C * OH * OW;
    if (i >= tot) return;
    int ox = i % OW; int t = i / OW;
    int oy = t % OH; t /= OH;
    int c = t % C;   int b = t / C;
    const int TH = OH + 1, TW = OW + 1;
    const float k1[4] = {0.125f, 0.375f, 0.375f, 0.125f};
    float acc = 0.f;
    #pragma unroll
    for (int ky = 0; ky < 4; ++ky) {
        int iy = oy + ky - 1;
        if (iy < 0 || iy >= TH) continue;
        #pragma unroll
        for (int kx = 0; kx < 4; ++kx) {
            int ix = ox + kx - 1;
            if (ix < 0 || ix >= TW) continue;
            acc += k1[ky] * k1[kx] * T[((b * C + c) * TH + iy) * TW + ix];
        }
    }
    acc = acc * 4.f + nw[0] * noise[(b * OH + oy) * OW + ox] + ab[c];
    out[i] = lrelu_g(acc);
}

// ---------------------------------------------------------------------------
// to_rgb: 1x1 modulated conv (no demod) + bias, plus up-blurred skip add.
// ---------------------------------------------------------------------------
__global__ void k_torgb(const float* __restrict__ x, const float* __restrict__ w,
                        const float* __restrict__ s, const float* __restrict__ bias,
                        const float* __restrict__ skip, float* __restrict__ out,
                        int cin, int OH, int OW) {
    int i = blockIdx.x * blockDim.x + threadIdx.x;
    int tot = 8 * 3 * OH * OW;
    if (i >= tot) return;
    int ox = i % OW; int t = i / OW;
    int oy = t % OH; t /= OH;
    int co = t % 3;  int b = t / 3;
    const float scale = rsqrtf((float)cin);
    float acc = 0.f;
    for (int ci = 0; ci < cin; ++ci)
        acc += w[co * cin + ci] * s[b * cin + ci] *
               x[((b * cin + ci) * OH + oy) * OW + ox];
    acc = acc * scale + bias[co];
    if (skip) {
        const int SH = OH >> 1, SW = OW >> 1;
        const float k1[4] = {0.125f, 0.375f, 0.375f, 0.125f};
        #pragma unroll
        for (int ky = 0; ky < 4; ++ky) {
            int ty = oy + ky - 2;
            if (ty < 0 || (ty & 1)) continue;
            int iy = ty >> 1; if (iy >= SH) continue;
            #pragma unroll
            for (int kx = 0; kx < 4; ++kx) {
                int tx = ox + kx - 2;
                if (tx < 0 || (tx & 1)) continue;
                int ix = tx >> 1; if (ix >= SW) continue;
                acc += 4.f * k1[ky] * k1[kx] *
                       skip[((b * 3 + co) * SH + iy) * SW + ix];
            }
        }
    }
    out[((b * 3 + co) * OH + oy) * OW + ox] = acc;
}

// ===========================================================================
// Host orchestration. d_in flattening assumed in dict insertion order:
//   0: z | 1..16: mapping (w,b)x8 | 17: const | 18..22: conv1 (w,mw,mb,nw,ab)
//   23..26: to_rgb1 (w,mw,mb,bias) | 27+5j: convs[j] | 87+4r: to_rgbs[r]
//   111+n: noises[n]
// ===========================================================================
extern "C" void kernel_launch(void* const* d_in, const int* in_sizes, int n_in,
                              void* d_out, int out_size, void* d_ws, size_t ws_size,
                              hipStream_t stream) {
    (void)in_sizes; (void)n_in; (void)out_size; (void)ws_size;
    auto in = [&](int i) { return (const float*)d_in[i]; };
    float* ws = (float*)d_ws;
    size_t off = 0;
    auto alloc = [&](size_t n) { float* p = ws + off; off += n; return p; };

    float* LAT0 = alloc(8 * 512);
    float* LAT1 = alloc(8 * 512);
    float* SB   = alloc(8 * 512);
    float* DB   = alloc(8 * 512);
    float* SRGB = alloc(8 * 512);
    _Float16* PW = (_Float16*)alloc(512 * 4608 / 2 + 64);   // packed f16 weights
    const size_t ACTSZ = 8ull * 128 * 257 * 129 + 1024;     // largest tensor
    float* B0  = alloc(ACTSZ);
    float* B1  = alloc(ACTSZ);
    float* B2  = alloc(ACTSZ);
    float* TMP = alloc(ACTSZ);
    float* SK0 = alloc(8 * 3 * 256 * 128);
    float* SK1 = alloc(8 * 3 * 256 * 128);

    // ---- mapping network ----
    k_pixelnorm<<<8, 512, 0, stream>>>(in(0), LAT0);
    float* lc = LAT0; float* ln = LAT1;
    for (int i = 0; i < 8; ++i) {
        k_map_layer<<<1, 512, 0, stream>>>(lc, in(1 + 2 * i), in(2 + 2 * i), ln);
        float* t = lc; lc = ln; ln = t;
    }
    const float* latent = lc;

    auto style = [&](const float* mw, const float* mb, float* sout, int cin) {
        k_style<<<(8 * cin + 255) / 256, 256, 0, stream>>>(latent, mw, mb, sout, cin);
    };
    auto pack = [&](const float* w, int cin, int cout, int flip) {
        int tot = cout * cin * 9;
        k_pack_w<<<(tot + 255) / 256, 256, 0, stream>>>(w, PW, cin, cout, flip);
    };

    // ---- conv1 on broadcast const [1,512,4,2], then to_rgb1 ----
    {
        const int cin = 512, cout = 512, H = 4, W = 2;
        style(in(19), in(20), SB, cin);
        k_demod<<<(8 * cout + 255) / 256, 256, 0, stream>>>(in(18), SB, DB, cin, cout);
        pack(in(18), cin, cout, 0);
        int N = 8 * H * W;
        dim3 g(cout / 64, (N + 63) / 64);
        k_conv_wmma<<<g, 256, 0, stream>>>(in(17), PW, SB, DB, in(111), in(21),
                                           in(22), B0, cin, cout, H, W, H, W,
                                           /*xbs=*/0, /*up=*/0, /*fuse=*/1);
        style(in(24), in(25), SRGB, cin);
        int tot = 8 * 3 * H * W;
        k_torgb<<<(tot + 255) / 256, 256, 0, stream>>>(B0, in(23), SRGB, in(26),
                                                       nullptr, SK0, cin, H, W);
    }

    float* act = B0; float* f1 = B1; float* f2 = B2;
    float* skip = SK0; float* skipN = SK1;
    int H = 4, Wd = 2, cin = 512;
    const int CH[6] = {512, 512, 512, 512, 256, 128};

    for (int i = 0; i < 6; ++i) {
        const int cout = CH[i];
        const int OH = 2 * H, OW = 2 * Wd;

        // upsample conv: convs[2i] -> TMP raw (2H+1)x(2W+1) -> blur+act -> f1
        {
            int base = 27 + 5 * (2 * i);
            style(in(base + 1), in(base + 2), SB, cin);
            k_demod<<<(8 * cout + 255) / 256, 256, 0, stream>>>(in(base), SB, DB, cin, cout);
            pack(in(base), cin, cout, 1);                  // pre-flipped kernel
            int TH = 2 * H + 1, TW = 2 * Wd + 1;
            int N = 8 * TH * TW;
            dim3 g(cout / 64, (N + 63) / 64);
            k_conv_wmma<<<g, 256, 0, stream>>>(act, PW, SB, DB, in(112 + 2 * i),
                                               in(base + 3), in(base + 4), TMP,
                                               cin, cout, H, Wd, TH, TW,
                                               cin * H * Wd, /*up=*/1, /*fuse=*/0);
            int tot = 8 * cout * OH * OW;
            k_blur_act<<<(tot + 255) / 256, 256, 0, stream>>>(TMP, in(112 + 2 * i),
                                                              in(base + 3), in(base + 4),
                                                              f1, cout, OH, OW);
        }
        // regular conv: convs[2i+1] on f1 -> f2 (fused noise/bias/lrelu)
        {
            int base = 27 + 5 * (2 * i + 1);
            style(in(base + 1), in(base + 2), SB, cout);
            k_demod<<<(8 * cout + 255) / 256, 256, 0, stream>>>(in(base), SB, DB, cout, cout);
            pack(in(base), cout, cout, 0);
            int N = 8 * OH * OW;
            dim3 g(cout / 64, (N + 63) / 64);
            k_conv_wmma<<<g, 256, 0, stream>>>(f1, PW, SB, DB, in(113 + 2 * i),
                                               in(base + 3), in(base + 4), f2,
                                               cout, cout, OH, OW, OH, OW,
                                               cout * OH * OW, /*up=*/0, /*fuse=*/1);
        }
        // to_rgb with up-blurred skip; final iteration writes d_out
        {
            int base = 87 + 4 * i;
            style(in(base + 1), in(base + 2), SRGB, cout);
            float* dst = (i == 5) ? (float*)d_out : skipN;
            int tot = 8 * 3 * OH * OW;
            k_torgb<<<(tot + 255) / 256, 256, 0, stream>>>(f2, in(base), SRGB,
                                                           in(base + 3), skip, dst,
                                                           cout, OH, OW);
            float* t = skip; skip = skipN; skipN = t;
        }
        // rotate: new current activation = f2; old act becomes a free buffer
        float* oldAct = act;
        act = f2; f2 = oldAct;
        cin = cout; H = OH; Wd = OW;
    }
}
